// ContrastiveLoss_70085276336663
// MI455X (gfx1250) — matrix operand order repaired and checked by reading, not compiled, not run on hardware
//
#include <hip/hip_runtime.h>
#include <hip/hip_bf16.h>

// ---------------------------------------------------------------------------
// NT-Xent contrastive loss, fused.  B=4096, D=1024, TEMP=0.5.
//   ws layout: reps_f16[8192*1024] | pos[8192] f32 | denom[8192] f32 | partials[32] f32
//   out layout: [loss, loss_partial[8192], num]  (8194 floats)
// ---------------------------------------------------------------------------

#define BROWS 4096
#define DDIM  1024
#define NROWS 8192   // 2*B

typedef __attribute__((ext_vector_type(16))) _Float16     v16h;
typedef __attribute__((ext_vector_type(8)))  _Float16     v8h;
typedef __attribute__((ext_vector_type(8)))  float        v8f;
typedef __attribute__((ext_vector_type(4)))  unsigned int v4u;
typedef __attribute__((ext_vector_type(8)))  int          v8i;
typedef __attribute__((ext_vector_type(4)))  int          v4i;

union V16 { v16h v; v8h h[2]; };

// ---------------------------------------------------------------------------
// Kernel 1: per-row norms, exact positives in f32, emit normalized rows as f16.
// ---------------------------------------------------------------------------
__global__ __launch_bounds__(256) void nk_norm_pos(
    const float* __restrict__ emb_i, const float* __restrict__ emb_j,
    _Float16* __restrict__ reps, float* __restrict__ pos) {
  const int r = blockIdx.x;
  const int t = threadIdx.x;
  const float* xi = emb_i + (size_t)r * DDIM;
  const float* xj = emb_j + (size_t)r * DDIM;

  float vi[4], vj[4];
  float si = 0.f, sj = 0.f, sd = 0.f;
#pragma unroll
  for (int k = 0; k < 4; ++k) {
    int c = t + k * 256;
    vi[k] = xi[c]; vj[k] = xj[c];
    si += vi[k] * vi[k];
    sj += vj[k] * vj[k];
    sd += vi[k] * vj[k];
  }
#pragma unroll
  for (int o = 16; o > 0; o >>= 1) {
    si += __shfl_down(si, o, 32);
    sj += __shfl_down(sj, o, 32);
    sd += __shfl_down(sd, o, 32);
  }
  __shared__ float rs[3][8];
  __shared__ float bc[2];
  const int wave = t >> 5, lane = t & 31;
  if (lane == 0) { rs[0][wave] = si; rs[1][wave] = sj; rs[2][wave] = sd; }
  __syncthreads();
  if (t == 0) {
    float a = 0.f, b = 0.f, d = 0.f;
#pragma unroll
    for (int w = 0; w < 8; ++w) { a += rs[0][w]; b += rs[1][w]; d += rs[2][w]; }
    float ni = fmaxf(sqrtf(a), 1e-12f);
    float nj = fmaxf(sqrtf(b), 1e-12f);
    float ii = 1.0f / ni, ij = 1.0f / nj;
    bc[0] = ii; bc[1] = ij;
    float p = d * ii * ij;          // exact cosine of the positive pair
    pos[r] = p;
    pos[r + BROWS] = p;
  }
  __syncthreads();
  const float ii = bc[0], ij = bc[1];
  _Float16* ri = reps + (size_t)r * DDIM;
  _Float16* rj = reps + (size_t)(r + BROWS) * DDIM;
#pragma unroll
  for (int k = 0; k < 4; ++k) {
    int c = t + k * 256;
    ri[c] = (_Float16)(vi[k] * ii);
    rj[c] = (_Float16)(vj[k] * ij);
  }
}

// ---------------------------------------------------------------------------
// Async staging of one B tile (128 cols x 32 K, f16) into an LDS buffer via
// GLOBAL_LOAD_ASYNC_TO_LDS_B128 (ASYNCcnt-tracked, no VGPR round trip).
// 512 16-byte chunks; 2 per thread -> 2 async instructions per wave per tile.
// ---------------------------------------------------------------------------
__device__ __forceinline__ void issue_b_tile(const _Float16* __restrict__ reps,
                                             _Float16* BsBuf, int colBase,
                                             int kc, int tid) {
#pragma unroll
  for (int i = 0; i < 2; ++i) {
    int ch = tid + i * 256;          // 0..511
    int cc = ch >> 2;                // 4 chunks per column
    int kk = (ch & 3) * 8;
    unsigned lds = (unsigned)(unsigned long long)&BsBuf[cc * 32 + kk];
    unsigned long long ga = (unsigned long long)
        &reps[(size_t)(colBase + cc) * DDIM + kc + kk];
    asm volatile("global_load_async_to_lds_b128 %0, %1, off"
                 :: "v"(lds), "v"(ga) : "memory");
  }
}

// ---------------------------------------------------------------------------
// Kernel 2: fused sim = reps@reps^T with online denominator accumulation.
// Grid: 128 blocks x 256 threads (8 waves).  Block owns 64 rows.
//   A panel 64x1024 f16 (128KB) staged once via TDM tensor_load_to_lds.
//   B tile 128x32 f16, double-buffered (16KB), staged via async-to-LDS with
//   one-tile lookahead overlapping the WMMA compute.
// ---------------------------------------------------------------------------
__global__ __launch_bounds__(256) void nk_gemm_denom(
    const _Float16* __restrict__ reps, float* __restrict__ denom) {
  __shared__ __align__(16) _Float16 As[64 * DDIM];     // [row][k]
  __shared__ __align__(16) _Float16 Bs[2][128 * 32];   // [buf][col][k]
  __shared__ float denomS[64];

  const int tid  = threadIdx.x;
  const int lane = tid & 31;
  const int wave = tid >> 5;
  const int mstrip  = wave & 3;
  const int colHalf = wave >> 2;
  const int l15 = lane & 15;
  const int hi  = lane >> 4;               // 0: lanes 0-15, 1: lanes 16-31
  const int rowBase = blockIdx.x * 64;

  if (tid < 64) denomS[tid] = 0.0f;

  // Prefetch the very first B tile while the A panel streams in.
  issue_b_tile(reps, &Bs[0][0], 0, 0, tid);

  // ---- Stage full A panel: 64 rows x 1024 K --------------------------------
#if __has_builtin(__builtin_amdgcn_tensor_load_to_lds) && \
    __has_builtin(__builtin_amdgcn_s_wait_tensorcnt)
  if (wave == 0) {
    // Tensor DMA descriptor (D#), 2D: tensor 1024x8192 f16 (stride 1024),
    // tile 1024x64 starting at reps[rowBase][0], destination = As.
    unsigned ldsA = (unsigned)(unsigned long long)&As[0];
    unsigned long long ga =
        (unsigned long long)(reps + (size_t)rowBase * DDIM);
    v4u g0;
    g0[0] = 1u;                                        // count=1, user mode
    g0[1] = ldsA;                                      // lds_addr
    g0[2] = (unsigned)ga;                              // global_addr[31:0]
    g0[3] = (unsigned)((ga >> 32) & 0x01FFFFFFu) | (2u << 30);  // addr hi | type=2
    v8i g1;
    g1[0] = (int)(1u << 16);                           // data_size=1 (2 bytes)
    g1[1] = (int)((DDIM & 0xFFFFu) << 16);             // tensor_dim0[15:0]
    g1[2] = (int)((DDIM >> 16) | ((NROWS & 0xFFFFu) << 16));   // dim0 hi|dim1 lo
    g1[3] = (int)((NROWS >> 16) | (DDIM << 16));       // dim1 hi | tile_dim0=1024
    g1[4] = 64;                                        // tile_dim1=64, tile_dim2=0
    g1[5] = DDIM;                                      // tensor_dim0_stride lo32
    g1[6] = 0;                                         // stride hi | dim1_stride lo
    g1[7] = 0;
    v4i gz4 = {0, 0, 0, 0};
    v8i gz8 = {0, 0, 0, 0, 0, 0, 0, 0};
    __builtin_amdgcn_tensor_load_to_lds(g0, g1, gz4, gz4, gz8, 0);
    __builtin_amdgcn_s_wait_tensorcnt(0);
  }
#else
  {
    const _Float16* ra = reps + (size_t)rowBase * DDIM;
#pragma unroll
    for (int i = 0; i < 32; ++i) {
      int ch = tid + i * 256;              // 0..8191
      int r  = ch >> 7;                    // 128 chunks per row
      int kk = (ch & 127) * 8;
      *(v8h*)&As[r * DDIM + kk] = *(const v8h*)&ra[(size_t)r * DDIM + kk];
    }
  }
#endif
  __syncthreads();

  const int aRow = mstrip * 16 + l15;      // A-fragment row in LDS panel
  float part[8];
#pragma unroll
  for (int v = 0; v < 8; ++v) part[v] = 0.0f;

  for (int colBase = 0; colBase < NROWS; colBase += 128) {
    v8f c0 = {}, c1 = {}, c2 = {}, c3 = {};

    for (int kc = 0; kc < DDIM; kc += 32) {
      const int buf = (kc >> 5) & 1;

      // Look-ahead: queue the next tile into the other buffer, then wait for
      // this wave's previous-tile asyncs (in-order completion => <=2 left).
      int nCol, nKc;
      bool has_next;
      if (kc + 32 < DDIM)            { nCol = colBase;       nKc = kc + 32; has_next = true; }
      else if (colBase + 128 < NROWS){ nCol = colBase + 128; nKc = 0;       has_next = true; }
      else                           { nCol = 0; nKc = 0;    has_next = false; }

      if (has_next) {
        issue_b_tile(reps, &Bs[buf ^ 1][0], nCol, nKc, tid);
        asm volatile("s_wait_asynccnt 0x2" ::: "memory");
      } else {
        asm volatile("s_wait_asynccnt 0x0" ::: "memory");
      }
      __syncthreads();               // current tile visible to all waves

      // A fragment: lanes 0-15 K{kc..+15},
      // lanes 16-31 K{kc+16..+31}
      V16 a;
      a.h[0] = *(const v8h*)&As[aRow * DDIM + kc + hi * 16];
      a.h[1] = *(const v8h*)&As[aRow * DDIM + kc + hi * 16 + 8];

      // B fragments: lane -> N = l15; K range hi*16..hi*16+15 of this chunk.
      const _Float16* Bc = &Bs[buf][0];
      const int bK = hi * 16;
      const int cb = (colHalf * 64 + l15) * 32;
      V16 b0, b1, b2, b3;
      b0.h[0] = *(const v8h*)&Bc[cb +  0 * 512 + bK];
      b0.h[1] = *(const v8h*)&Bc[cb +  0 * 512 + bK + 8];
      b1.h[0] = *(const v8h*)&Bc[cb +  1 * 512 + bK];
      b1.h[1] = *(const v8h*)&Bc[cb +  1 * 512 + bK + 8];
      b2.h[0] = *(const v8h*)&Bc[cb +  2 * 512 + bK];
      b2.h[1] = *(const v8h*)&Bc[cb +  2 * 512 + bK + 8];
      b3.h[0] = *(const v8h*)&Bc[cb +  3 * 512 + bK];
      b3.h[1] = *(const v8h*)&Bc[cb +  3 * 512 + bK + 8];

      c0 = __builtin_amdgcn_wmma_f32_16x16x32_f16(false, a.v, false, b0.v,
                                                  (short)0, c0, false, false);
      c1 = __builtin_amdgcn_wmma_f32_16x16x32_f16(false, a.v, false, b1.v,
                                                  (short)0, c1, false, false);
      c2 = __builtin_amdgcn_wmma_f32_16x16x32_f16(false, a.v, false, b2.v,
                                                  (short)0, c2, false, false);
      c3 = __builtin_amdgcn_wmma_f32_16x16x32_f16(false, a.v, false, b3.v,
                                                  (short)0, c3, false, false);
      __syncthreads();               // all waves done reading buf before reuse
    }

    // Post-process the 16x64 wave tile: exp(sim/0.5), mask diagonal, row-accum.
    // C layout: lane -> N = l15; VGPR v -> M = hi*8 + v.
    const int rowG = rowBase + mstrip * 16 + hi * 8;
    const int colT = colBase + colHalf * 64 + l15;
#pragma unroll
    for (int v = 0; v < 8; ++v) {
      float e0 = __expf(2.0f * c0[v]);
      float e1 = __expf(2.0f * c1[v]);
      float e2 = __expf(2.0f * c2[v]);
      float e3 = __expf(2.0f * c3[v]);
      const int rg = rowG + v;
      if (colT      == rg) e0 = 0.0f;
      if (colT + 16 == rg) e1 = 0.0f;
      if (colT + 32 == rg) e2 = 0.0f;
      if (colT + 48 == rg) e3 = 0.0f;
      part[v] += (e0 + e1) + (e2 + e3);
    }
  }

  // Reduce each row sum across the 16 lanes of each half-wave.
#pragma unroll
  for (int v = 0; v < 8; ++v) {
#pragma unroll
    for (int o = 1; o < 16; o <<= 1) part[v] += __shfl_xor(part[v], o, 16);
  }
  if (l15 == 0) {
#pragma unroll
    for (int v = 0; v < 8; ++v)
      atomicAdd(&denomS[mstrip * 16 + hi * 8 + v], part[v]);
  }
  __syncthreads();
  if (tid < 64) denom[rowBase + tid] = denomS[tid];
}

// ---------------------------------------------------------------------------
// Kernel 3: loss_partial[i] = -(pos[i]/T - log(denom[i])), block partial sums.
// ---------------------------------------------------------------------------
__global__ __launch_bounds__(256) void nk_loss_partial(
    const float* __restrict__ pos, const float* __restrict__ denom,
    float* __restrict__ out_lp, float* __restrict__ partials) {
  const int i = blockIdx.x * 256 + threadIdx.x;
  float lp = -(2.0f * pos[i] - __logf(denom[i]));
  out_lp[i] = lp;

  float s = lp;
#pragma unroll
  for (int o = 16; o > 0; o >>= 1) s += __shfl_down(s, o, 32);
  __shared__ float red[8];
  const int wave = threadIdx.x >> 5, lane = threadIdx.x & 31;
  if (lane == 0) red[wave] = s;
  __syncthreads();
  if (threadIdx.x == 0) {
    float t = 0.f;
#pragma unroll
    for (int w = 0; w < 8; ++w) t += red[w];
    partials[blockIdx.x] = t;
  }
}

// ---------------------------------------------------------------------------
// Kernel 4: final scalar reduction (deterministic, no global atomics).
// ---------------------------------------------------------------------------
__global__ void nk_finalize(const float* __restrict__ partials,
                            float* __restrict__ out) {
  if (threadIdx.x == 0) {
    float t = 0.f;
    for (int i = 0; i < 32; ++i) t += partials[i];
    const float loss = t / (float)NROWS;
    out[0] = loss;          // loss
    out[1 + NROWS] = loss;  // num (== loss numerically)
  }
}

// ---------------------------------------------------------------------------
extern "C" void kernel_launch(void* const* d_in, const int* in_sizes, int n_in,
                              void* d_out, int out_size, void* d_ws, size_t ws_size,
                              hipStream_t stream) {
  const float* emb_i = (const float*)d_in[0];
  const float* emb_j = (const float*)d_in[1];
  float* out = (float*)d_out;

  char* ws = (char*)d_ws;
  _Float16* reps = (_Float16*)ws;                                 // 16 MB
  float* pos      = (float*)(ws + (size_t)NROWS * DDIM * 2);      // 32 KB
  float* denom    = pos + NROWS;                                  // 32 KB
  float* partials = denom + NROWS;                                // 128 B

  nk_norm_pos<<<BROWS, 256, 0, stream>>>(emb_i, emb_j, reps, pos);
  nk_gemm_denom<<<NROWS / 64, 256, 0, stream>>>(reps, denom);
  nk_loss_partial<<<NROWS / 256, 256, 0, stream>>>(pos, denom, out + 1, partials);
  nk_finalize<<<1, 64, 0, stream>>>(partials, out);
}
